// ValenceModel_64673617543725
// MI455X (gfx1250) — matrix operand order repaired and checked by reading, not compiled, not run on hardware
//
#include <hip/hip_runtime.h>

typedef _Float16 f16;
typedef __attribute__((ext_vector_type(16))) _Float16 v16h;
typedef __attribute__((ext_vector_type(8)))  float    v8f;

union ABFrag { uint4 u[2]; v16h h; };

struct Perms { int p[3][4]; };

__device__ __forceinline__ v8f wmma_f16(const ABFrag& a, const ABFrag& b, v8f c) {
  return __builtin_amdgcn_wmma_f32_16x16x32_f16(false, a.h, false, b.h, (short)0, c, false, false);
}

// ---------------- prep kernels ----------------
__global__ void cvt_f16_kernel(const float* __restrict__ s, f16* __restrict__ d, int n) {
  int i = blockIdx.x * 256 + threadIdx.x;
  if (i < n) d[i] = (f16)s[i];
}

// W [K,128] f32  ->  WT [128,K] f16   (column of W becomes contiguous row)
__global__ void transpose_w_kernel(const float* __restrict__ W, f16* __restrict__ WT, int K) {
  int i = blockIdx.x * 256 + threadIdx.x;
  if (i < K * 128) {
    int k = i >> 7, n = i & 127;
    WT[(size_t)n * K + k] = (f16)W[i];
  }
}

// ---------------- node representation ----------------
// nodeRep[g,:] = relu(embed[elements[g]] @ node_W + node_b), stored f16
__global__ void node_kernel(const f16* __restrict__ embedF16,
                            const int* __restrict__ elements,
                            const f16* __restrict__ nodeWT,   // [128,128] f16 (W^T)
                            const float* __restrict__ node_b,
                            f16* __restrict__ nodeRep,
                            int R) {
  __shared__ __align__(16) f16 Xs[16 * 128];  // A tile, then reused as f16 output stage
  __shared__ int sIdx[16];
  const int tid  = threadIdx.x;
  const int lane = tid & 31;
  const int wave = tid >> 5;                  // 0..7 -> output columns [16w,16w+16)
  const int rowbase = blockIdx.x * 16;

  if (tid < 16) {
    int g = rowbase + tid; if (g >= R) g = R - 1;
    sIdx[tid] = elements[g];
  }
  __syncthreads();
  { // gather 16 rows x 256B, one uint4 per thread
    int r = tid >> 4, ch = tid & 15;
    const uint4* src = reinterpret_cast<const uint4*>(embedF16 + (size_t)sIdx[r] * 128) + ch;
    reinterpret_cast<uint4*>(Xs + r * 128)[ch] = *src;
  }
  __syncthreads();

  const int m    = lane & 15;
  const int half = lane >> 4;
  const int aoff = half * 8;                  // A: K offset {0,8}
  const int n    = wave * 16 + m;             // output column
  const int boff = half * 8;                  // B: K offset {0,8}

  v8f acc = {};
#pragma unroll
  for (int kb = 0; kb < 128; kb += 32) {
    ABFrag a, b;
    const uint4* pa = reinterpret_cast<const uint4*>(Xs + m * 128 + kb + aoff);
    a.u[0] = pa[0]; a.u[1] = pa[2];
    const uint4* pb = reinterpret_cast<const uint4*>(nodeWT + (size_t)n * 128 + kb + boff);
    b.u[0] = pb[0]; b.u[1] = pb[2];
    acc = wmma_f16(a, b, acc);
  }
  const float bias = node_b[n];
  const int m0 = half * 8;                    // C: row = r + 8*(lane>=16)

  // stage relu+f16 result in LDS (Xs reuse), then coalesced 16B stores
  __syncthreads();
#pragma unroll
  for (int r = 0; r < 8; ++r)
    Xs[(m0 + r) * 128 + n] = (f16)fmaxf(acc[r] + bias, 0.0f);
  __syncthreads();
  {
    int r = tid >> 4, ch = tid & 15;
    int g = rowbase + r;
    if (g < R)
      reinterpret_cast<uint4*>(nodeRep + (size_t)g * 128)[ch] =
          reinterpret_cast<const uint4*>(Xs + r * 128)[ch];
  }
}

// ---------------- generic symmetry-pooled interaction MLP ----------------
// 32 rows per block (2 M-tiles per wave, B-fragment reused for 2 WMMAs)
// out[g,:] = (sum_perm relu(concat(nodeRep[idx]) @ W1 + b1)) @ W2 + NPERM*b2
template<int ARITY, int NPERM, int ODIM, int NCOLS>
__global__ void interaction_kernel(const f16* __restrict__ nodeRep,
                                   const int* __restrict__ indices,  // [R,NCOLS] or nullptr
                                   Perms perms,
                                   const f16* __restrict__ W1T,      // [128, K] f16 (W1^T)
                                   const float* __restrict__ b1,     // [128]
                                   const float* __restrict__ W2,     // [128, ODIM]
                                   const float* __restrict__ b2,     // [ODIM]
                                   float* __restrict__ out,          // [R, ODIM]
                                   int R) {
  constexpr int K = ARITY * 128;
  __shared__ __align__(16) f16   Xs[32 * K];      // up to 32 KB
  __shared__ __align__(16) float Hs[32][128];     // 16 KB
  __shared__ int sIdx[32 * ARITY];

  const int tid  = threadIdx.x;
  const int lane = tid & 31;
  const int wave = tid >> 5;
  const int rowbase = blockIdx.x * 32;

  const int m    = lane & 15;
  const int half = lane >> 4;
  const int aoff = half * 8;
  const int n    = wave * 16 + m;
  const int boff = half * 8;

  v8f hsum0 = {}, hsum1 = {};

  for (int pi = 0; pi < NPERM; ++pi) {
    __syncthreads();                              // Xs free before overwrite
    if (tid < 32 * ARITY) {
      int r = tid / ARITY, s = tid % ARITY;
      int g = rowbase + r; if (g >= R) g = R - 1;
      sIdx[tid] = indices ? indices[(size_t)g * NCOLS + perms.p[pi][s]] : g;
    }
    __syncthreads();
    // gather X tile via async global->LDS (ASYNCcnt path, no VGPR round-trip)
    for (int c = tid; c < 32 * ARITY * 16; c += 256) {
      int r   = c / (ARITY * 16);
      int rem = c - r * (ARITY * 16);
      int s   = rem >> 4;
      int ch  = rem & 15;
      int idx = sIdx[r * ARITY + s];
      const uint4* src = reinterpret_cast<const uint4*>(nodeRep + (size_t)idx * 128) + ch;
      // flat->LDS aperture: LDS byte address = low 32 bits of the shared pointer
      unsigned ldsAddr = (unsigned)(uintptr_t)(Xs + r * K + s * 128) + (unsigned)ch * 16u;
      asm volatile("global_load_async_to_lds_b128 %0, %1, off"
                   :: "v"(ldsAddr), "v"(src) : "memory");
    }
    asm volatile("s_wait_asynccnt 0" ::: "memory");
    __syncthreads();

    v8f acc0 = {}, acc1 = {};
#pragma unroll
    for (int kb = 0; kb < K; kb += 32) {
      ABFrag b;
      const uint4* pb = reinterpret_cast<const uint4*>(W1T + (size_t)n * K + kb + boff);
      b.u[0] = pb[0]; b.u[1] = pb[2];
      ABFrag a0, a1;
      const uint4* pa0 = reinterpret_cast<const uint4*>(Xs + m * K + kb + aoff);
      a0.u[0] = pa0[0]; a0.u[1] = pa0[2];
      const uint4* pa1 = reinterpret_cast<const uint4*>(Xs + (16 + m) * K + kb + aoff);
      a1.u[0] = pa1[0]; a1.u[1] = pa1[2];
      acc0 = wmma_f16(a0, b, acc0);
      acc1 = wmma_f16(a1, b, acc1);
    }
    const float bias = b1[n];
#pragma unroll
    for (int r = 0; r < 8; ++r) {
      hsum0[r] += fmaxf(acc0[r] + bias, 0.0f);
      hsum1[r] += fmaxf(acc1[r] + bias, 0.0f);
    }
  }

  // layer 2 via LDS H tile
  const int m0 = half * 8;
#pragma unroll
  for (int r = 0; r < 8; ++r) {
    Hs[m0 + r][n]      = hsum0[r];
    Hs[16 + m0 + r][n] = hsum1[r];
  }
  __syncthreads();

  for (int t = tid; t < 32 * ODIM; t += 256) {
    int r = t / ODIM, j = t - r * ODIM;
    int g = rowbase + r;
    if (g < R) {
      float a2 = (float)NPERM * b2[j];
#pragma unroll 8
      for (int k = 0; k < 128; ++k) a2 += Hs[r][k] * W2[k * ODIM + j];
      out[(size_t)g * ODIM + j] = a2;
    }
  }
}

// ---------------- launch ----------------
extern "C" void kernel_launch(void* const* d_in, const int* in_sizes, int n_in,
                              void* d_out, int out_size, void* d_ws, size_t ws_size,
                              hipStream_t stream) {
  const int*   elements  = (const int*)  d_in[0];
  const int*   bonds     = (const int*)  d_in[1];
  const int*   angles    = (const int*)  d_in[2];
  const int*   propers   = (const int*)  d_in[3];
  const int*   impropers = (const int*)  d_in[4];
  const float* embed     = (const float*)d_in[5];
  const float* node_W    = (const float*)d_in[6];
  const float* node_b    = (const float*)d_in[7];
  const float* atom_W1 = (const float*)d_in[8],  *atom_b1 = (const float*)d_in[9];
  const float* atom_W2 = (const float*)d_in[10], *atom_b2 = (const float*)d_in[11];
  const float* bond_W1 = (const float*)d_in[12], *bond_b1 = (const float*)d_in[13];
  const float* bond_W2 = (const float*)d_in[14], *bond_b2 = (const float*)d_in[15];
  const float* angle_W1 = (const float*)d_in[16], *angle_b1 = (const float*)d_in[17];
  const float* angle_W2 = (const float*)d_in[18], *angle_b2 = (const float*)d_in[19];
  const float* proper_W1 = (const float*)d_in[20], *proper_b1 = (const float*)d_in[21];
  const float* proper_W2 = (const float*)d_in[22], *proper_b2 = (const float*)d_in[23];
  const float* improper_W1 = (const float*)d_in[24], *improper_b1 = (const float*)d_in[25];
  const float* improper_W2 = (const float*)d_in[26], *improper_b2 = (const float*)d_in[27];

  const int N  = in_sizes[0];
  const int NB = in_sizes[1] / 2;
  const int NA = in_sizes[2] / 3;
  const int NP = in_sizes[3] / 4;
  const int NI = in_sizes[4] / 5;
  const int EMB = in_sizes[5];           // VOCAB*128

  // workspace layout (f16), ~26.2 MB total
  f16* nodeRep     = (f16*)d_ws;                               // N*128
  f16* embedF16    = nodeRep  + (size_t)N * 128;               // EMB
  f16* nodeWT      = embedF16 + EMB;                           // 128*128
  f16* atomW1T     = nodeWT     + 128 * 128;                   // 128*128
  f16* bondW1T     = atomW1T    + 128 * 128;                   // 128*256
  f16* angleW1T    = bondW1T    + 128 * 256;                   // 128*384
  f16* properW1T   = angleW1T   + 128 * 384;                   // 128*512
  f16* improperW1T = properW1T  + 128 * 512;                   // 128*512

  cvt_f16_kernel<<<(EMB + 255) / 256, 256, 0, stream>>>(embed, embedF16, EMB);
  transpose_w_kernel<<<(128 * 128 + 255) / 256, 256, 0, stream>>>(node_W,      nodeWT,      128);
  transpose_w_kernel<<<(128 * 128 + 255) / 256, 256, 0, stream>>>(atom_W1,     atomW1T,     128);
  transpose_w_kernel<<<(256 * 128 + 255) / 256, 256, 0, stream>>>(bond_W1,     bondW1T,     256);
  transpose_w_kernel<<<(384 * 128 + 255) / 256, 256, 0, stream>>>(angle_W1,    angleW1T,    384);
  transpose_w_kernel<<<(512 * 128 + 255) / 256, 256, 0, stream>>>(proper_W1,   properW1T,   512);
  transpose_w_kernel<<<(512 * 128 + 255) / 256, 256, 0, stream>>>(improper_W1, improperW1T, 512);

  node_kernel<<<(N + 15) / 16, 256, 0, stream>>>(embedF16, elements, nodeWT, node_b, nodeRep, N);

  float* out          = (float*)d_out;
  float* atomsOut     = out;
  float* bondsOut     = atomsOut  + (size_t)N  * 2;
  float* anglesOut    = bondsOut  + (size_t)NB * 2;
  float* propersOut   = anglesOut + (size_t)NA * 2;
  float* impropersOut = propersOut + (size_t)NP * 6;

  Perms pAtom = {{{0,0,0,0},{0,0,0,0},{0,0,0,0}}};
  interaction_kernel<1,1,2,1><<<(N + 31) / 32, 256, 0, stream>>>(
      nodeRep, nullptr, pAtom, atomW1T, atom_b1, atom_W2, atom_b2, atomsOut, N);

  Perms pBond = {{{0,1,0,0},{1,0,0,0},{0,0,0,0}}};
  interaction_kernel<2,2,2,2><<<(NB + 31) / 32, 256, 0, stream>>>(
      nodeRep, bonds, pBond, bondW1T, bond_b1, bond_W2, bond_b2, bondsOut, NB);

  Perms pAngle = {{{0,1,2,0},{2,1,0,0},{0,0,0,0}}};
  interaction_kernel<3,2,2,3><<<(NA + 31) / 32, 256, 0, stream>>>(
      nodeRep, angles, pAngle, angleW1T, angle_b1, angle_W2, angle_b2, anglesOut, NA);

  Perms pProper = {{{0,1,2,3},{3,2,1,0},{0,0,0,0}}};
  interaction_kernel<4,2,6,4><<<(NP + 31) / 32, 256, 0, stream>>>(
      nodeRep, propers, pProper, properW1T, proper_b1, proper_W2, proper_b2, propersOut, NP);

  Perms pImp = {{{0,1,2,3},{0,3,4,2},{0,4,2,3}}};
  interaction_kernel<4,3,6,5><<<(NI + 31) / 32, 256, 0, stream>>>(
      nodeRep, impropers, pImp, improperW1T, improper_b1, improper_W2, improper_b2, impropersOut, NI);

  (void)n_in; (void)out_size; (void)ws_size;
}
